// Vrnn_48266842472915
// MI455X (gfx1250) — compile-verified, matrix operand order checked by reading, not run on hardware
//
#include <hip/hip_runtime.h>

// ---------------------------------------------------------------------------
// VRNN for MI455X (gfx1250, wave32, WMMA + async-tensor data path).
//  * GEMMs: v_wmma_f32_16x16x32_bf16, A/B K-panels staged in LDS via
//    global_load_async_to_lds_b128 (ASYNCcnt), double-buffered, consumed with
//    ds_load_b128. Panels are loaded once per block and shared by 8 waves.
//  * Weights converted once per launch to transposed bf16 (W^T [N,K]) so the
//    B-fragment LDS pattern matches the A-fragment pattern.
//  * Sequential scan: 12 small kernels per timestep; stream order = sync.
//    Weights stay L2-resident (192MB L2 >> 8MB bf16 weights).
//  * Dead computation skipped: prior branch (Wr*) and GRU states s2/s3 never
//    reach the outputs theta_mu/theta_sig/z.
// ---------------------------------------------------------------------------

typedef __bf16 bf16;
typedef __attribute__((ext_vector_type(16))) __bf16 v16bf;
typedef __attribute__((ext_vector_type(8)))  __bf16 v8bf;
typedef __attribute__((ext_vector_type(8)))  float   v8f;

__device__ __forceinline__ float softplusf(float x) {
  return fmaxf(x, 0.f) + log1pf(expf(-fabsf(x)));
}
__device__ __forceinline__ float sigmoidf(float x) {
  return 1.f / (1.f + expf(-x));
}

// ---------------------------------------------------------------------------
// Tiled bf16 WMMA GEMM with async-LDS double buffering:
//   D = act(A @ W + bias)
// A row-major bf16, optionally two K-segments (fused concat):
//   cols [0,kc0) from A0 (ld lda0), cols [kc0,Ktot) from A1 (ld lda1).
// W transposed: Wt [N,Ktot] row-major bf16.
// Block = 256 threads (8 waves) computes a 64x64 tile; wave computes 16x32.
// K stepped by 32; per stage each thread issues two async b128 loads
// (A panel 64x32, B panel 64x32 -> 4KB each, x2 buffers = 16KB LDS).
// Optional f32/bf16 outputs; optional row remap
//   orow = (r % rm_d1) * rm_d0 + r / rm_d1   (rm_d1 == 0 -> identity).
// Requires: M % 64 == 0, N % 64 == 0, Ktot % 32 == 0, kc0 % 32 == 0.
// ---------------------------------------------------------------------------
__global__ __launch_bounds__(256) void gemm_bf16_wmma(
    const bf16* __restrict__ A0, int lda0, int kc0,
    const bf16* __restrict__ A1, int lda1,
    const bf16* __restrict__ Wt,
    const float* __restrict__ bias,
    float* __restrict__ Df, int ldDf,
    bf16*  __restrict__ Db, int ldDb,
    int M, int N, int Ktot, int act, int rm_d0, int rm_d1)
{
  __shared__ __align__(16) bf16 sA[2][64 * 32];
  __shared__ __align__(16) bf16 sB[2][64 * 32];

  const int tid     = threadIdx.x;
  const int mblocks = M >> 6;
  const int rb      = blockIdx.x % mblocks;
  const int cb      = blockIdx.x / mblocks;
  const int row0    = rb << 6;
  const int col0    = cb << 6;
  const int wv      = tid >> 5;
  const int lane    = tid & 31;
  const int rt      = wv & 3;         // row-tile within block (0..3)
  const int cp      = wv >> 2;        // column half (0..1), 32 cols each
  const int r       = lane & 15;
  const int hs      = lane >> 4;      // K-group half select

  // async copy mapping: thread -> (row, 16B chunk) of a 64x32 panel
  const int arow = tid >> 2;          // 0..63
  const int achk = (tid & 3) << 3;    // element offset: 0,8,16,24

  // workgroup-relative LDS byte offsets (generic ptr low 32 bits = LDS offset)
  const unsigned sAoff = (unsigned)(unsigned long long)(&sA[0][0]);
  const unsigned sBoff = (unsigned)(unsigned long long)(&sB[0][0]);
  const unsigned panB  = 64 * 32 * 2; // bytes per panel buffer

  v8f acc0 = {}, acc1 = {};
  const int nk = Ktot >> 5;

#define ISSUE_STAGE(bufi, k0v)                                                 \
  do {                                                                         \
    const int _k0 = (k0v);                                                     \
    const bf16* _ap; int _kk;                                                  \
    if (_k0 < kc0) { _ap = A0 + (size_t)(row0 + arow) * lda0; _kk = _k0; }     \
    else           { _ap = A1 + (size_t)(row0 + arow) * lda1; _kk = _k0-kc0; } \
    const bf16* _ga = _ap + _kk + achk;                                        \
    unsigned _la = sAoff + (bufi) * panB + (unsigned)(arow * 32 + achk) * 2;   \
    asm volatile("global_load_async_to_lds_b128 %0, %1, off"                   \
                 :: "v"(_la), "v"(_ga) : "memory");                            \
    const bf16* _gb = Wt + (size_t)(col0 + arow) * Ktot + _k0 + achk;          \
    unsigned _lb = sBoff + (bufi) * panB + (unsigned)(arow * 32 + achk) * 2;   \
    asm volatile("global_load_async_to_lds_b128 %0, %1, off"                   \
                 :: "v"(_lb), "v"(_gb) : "memory");                            \
  } while (0)

  ISSUE_STAGE(0, 0);

  for (int s = 0; s < nk; ++s) {
    const int buf = s & 1;
    if (s + 1 < nk) {
      ISSUE_STAGE(buf ^ 1, (s + 1) << 5);
      asm volatile("s_wait_asynccnt 0x2" ::: "memory"); // older buffer landed
    } else {
      asm volatile("s_wait_asynccnt 0x0" ::: "memory");
    }
    __syncthreads();  // all waves' writes to buf visible

    // A fragment from LDS (ds_load_b128 x2), shared across both col tiles
    const bf16* pa = &sA[buf][(rt * 16 + r) * 32 + hs * 8];
    v8bf alo = *(const v8bf*)pa;
    v8bf ahi = *(const v8bf*)(pa + 16);
    v16bf af;
#pragma unroll
    for (int i = 0; i < 8; ++i) { af[i] = alo[i]; af[i + 8] = ahi[i]; }

    {
      const bf16* pb = &sB[buf][(cp * 32 + r) * 32 + hs * 8];
      v8bf blo = *(const v8bf*)pb;
      v8bf bhi = *(const v8bf*)(pb + 16);
      v16bf bfg;
#pragma unroll
      for (int i = 0; i < 8; ++i) { bfg[i] = blo[i]; bfg[i + 8] = bhi[i]; }
      acc0 = __builtin_amdgcn_wmma_f32_16x16x32_bf16(
          false, af, false, bfg, (short)0, acc0, false, false);
    }
    {
      const bf16* pb = &sB[buf][(cp * 32 + 16 + r) * 32 + hs * 8];
      v8bf blo = *(const v8bf*)pb;
      v8bf bhi = *(const v8bf*)(pb + 16);
      v16bf bfg;
#pragma unroll
      for (int i = 0; i < 8; ++i) { bfg[i] = blo[i]; bfg[i + 8] = bhi[i]; }
      acc1 = __builtin_amdgcn_wmma_f32_16x16x32_bf16(
          false, af, false, bfg, (short)0, acc1, false, false);
    }
    __syncthreads();  // reads of buf done before it is refilled next iter
  }
#undef ISSUE_STAGE

  // Epilogue: D element (m = v + 8*hs, n = lane&15) per C/D VGPR layout
#pragma unroll
  for (int c = 0; c < 2; ++c) {
    const v8f* accp = c ? &acc1 : &acc0;
    const int n  = col0 + cp * 32 + c * 16 + r;
    const float bb = bias ? bias[n] : 0.f;
#pragma unroll
    for (int v = 0; v < 8; ++v) {
      float y = (*accp)[v] + bb;
      if (act == 1)      y = fmaxf(y, 0.f);
      else if (act == 2) y = softplusf(y);
      const int rg   = row0 + rt * 16 + v + 8 * hs;
      const int orow = rm_d1 ? ((rg % rm_d1) * rm_d0 + rg / rm_d1) : rg;
      if (Df) Df[(size_t)orow * ldDf + n] = y;
      if (Db) Db[(size_t)orow * ldDb + n] = (bf16)y;
    }
  }
}

// f32 -> bf16 elementwise convert
__global__ void conv_bf16(const float* __restrict__ s, bf16* __restrict__ d, int n) {
  int i = blockIdx.x * blockDim.x + threadIdx.x;
  if (i < n) d[i] = (bf16)s[i];
}

// f32 [K,N] row-major -> bf16 transposed [N,K]
__global__ void wconv_t(const float* __restrict__ s, bf16* __restrict__ d, int K, int N) {
  int i = blockIdx.x * blockDim.x + threadIdx.x;
  if (i >= K * N) return;
  int k = i / N, n = i % N;
  d[(size_t)n * K + k] = (bf16)s[i];
}

__global__ void copyf(const float* __restrict__ s, float* __restrict__ d, int n) {
  int i = blockIdx.x * blockDim.x + threadIdx.x;
  if (i < n) d[i] = s[i];
}

// z_t = pmu + softplus(psig_pre) * eps[:,t,:]  (writes f32 to d_out, bf16 scratch)
__global__ void z_elem(const float* __restrict__ pz, const float* __restrict__ eps,
                       float* __restrict__ zout, bf16* __restrict__ zbuf, int t) {
  int i = blockIdx.x * blockDim.x + threadIdx.x;
  if (i >= 64 * 64) return;
  int b = i >> 6, c = i & 63;
  float mu  = pz[b * 128 + c];
  float sig = softplusf(pz[b * 128 + 64 + c]);
  float e   = eps[((size_t)b * 256 + t) * 64 + c];
  float z   = mu + sig * e;
  zout[((size_t)b * 256 + t) * 64 + c] = z;
  zbuf[i] = (bf16)z;
}

// Keras GRUCell (reset_after=True) elementwise combine given mx, mh (bias-included)
__global__ void gru_combine(const float* __restrict__ mx, const float* __restrict__ mh,
                            float* __restrict__ hf, bf16* __restrict__ hb,
                            bf16* __restrict__ sout) {
  int i = blockIdx.x * blockDim.x + threadIdx.x;
  if (i >= 64 * 512) return;
  int b = i >> 9, c = i & 511;
  const float* mxr = mx + (size_t)b * 1536;
  const float* mhr = mh + (size_t)b * 1536;
  float zg = sigmoidf(mxr[c] + mhr[c]);
  float rg = sigmoidf(mxr[512 + c] + mhr[512 + c]);
  float hh = tanhf(mxr[1024 + c] + rg * mhr[1024 + c]);
  float h  = hf[i];
  float hn = zg * h + (1.f - zg) * hh;
  hf[i] = hn;
  hb[i] = (bf16)hn;
  if (sout) sout[i] = (bf16)hn;
}

extern "C" void kernel_launch(void* const* d_in, const int* in_sizes, int n_in,
                              void* d_out, int out_size, void* d_ws, size_t ws_size,
                              hipStream_t stream) {
  (void)in_sizes; (void)n_in; (void)out_size; (void)ws_size;
  const int B = 64, T = 256, BT = B * T;

  const float* inX  = (const float*)d_in[0];
  const float* eps  = (const float*)d_in[1];
  const float* Wx1  = (const float*)d_in[2];  const float* bx1 = (const float*)d_in[3];
  const float* Wx2  = (const float*)d_in[4];  const float* bx2 = (const float*)d_in[5];
  const float* Wz1  = (const float*)d_in[6];  const float* bz1 = (const float*)d_in[7];
  const float* Wz2  = (const float*)d_in[8];  const float* bz2 = (const float*)d_in[9];
  const float* Wp1  = (const float*)d_in[10]; const float* bp1 = (const float*)d_in[11];
  const float* Wp2  = (const float*)d_in[12]; const float* bp2 = (const float*)d_in[13];
  const float* Wpmu = (const float*)d_in[14]; const float* bpmu = (const float*)d_in[15];
  const float* Wpsig= (const float*)d_in[16]; const float* bpsig= (const float*)d_in[17];
  // d_in[18..25]: prior branch (Wr*/br*) — dead w.r.t. outputs, skipped.
  const float* Wt1  = (const float*)d_in[26]; const float* bt1 = (const float*)d_in[27];
  const float* Wt2  = (const float*)d_in[28]; const float* bt2 = (const float*)d_in[29];
  const float* Wtmu = (const float*)d_in[30]; const float* btmu = (const float*)d_in[31];
  const float* Wtsig= (const float*)d_in[32]; const float* btsig= (const float*)d_in[33];
  const float* k1   = (const float*)d_in[34]; const float* rk1 = (const float*)d_in[35];
  const float* b1   = (const float*)d_in[36];
  const float* k2   = (const float*)d_in[37]; const float* rk2 = (const float*)d_in[38];
  const float* b2   = (const float*)d_in[39];

  // ---- workspace carve (256B aligned) ----
  size_t off = 0; char* base = (char*)d_ws;
  auto alloc = [&](size_t bytes) -> void* {
    void* p = base + off; off = (off + bytes + 255) & ~(size_t)255; return p;
  };
  bf16* Xin   = (bf16*)alloc((size_t)BT * 128 * 2);   // raw input, bf16
  bf16* x1b   = (bf16*)alloc((size_t)BT * 256 * 2);   // X-MLP hidden
  bf16* Xbf   = (bf16*)alloc((size_t)BT * 256 * 2);   // X transformed, [T,B,256]
  bf16* zpA   = (bf16*)alloc((size_t)BT * 256 * 2);   // zp, [T,B,256]
  bf16* SA    = (bf16*)alloc((size_t)BT * 512 * 2);   // s4 states, [T,B,512]
  bf16* th1   = (bf16*)alloc((size_t)BT * 256 * 2);
  bf16* th2   = (bf16*)alloc((size_t)BT * 256 * 2);
  // transposed bf16 weights
  bf16* Wx1t  = (bf16*)alloc((size_t)256 * 128 * 2);
  bf16* Wx2t  = (bf16*)alloc((size_t)256 * 256 * 2);
  bf16* Wz1t  = (bf16*)alloc((size_t)256 * 64  * 2);
  bf16* Wz2t  = (bf16*)alloc((size_t)256 * 256 * 2);
  bf16* Wp1t  = (bf16*)alloc((size_t)256 * 768 * 2);
  bf16* Wp2t  = (bf16*)alloc((size_t)256 * 256 * 2);
  bf16* Wpzt  = (bf16*)alloc((size_t)128 * 256 * 2);  // [mu;sig] stacked
  bf16* Wt1t  = (bf16*)alloc((size_t)256 * 768 * 2);
  bf16* Wt2t  = (bf16*)alloc((size_t)256 * 256 * 2);
  bf16* Wtmut = (bf16*)alloc((size_t)128 * 256 * 2);
  bf16* Wtsigt= (bf16*)alloc((size_t)128 * 256 * 2);
  bf16* k1t   = (bf16*)alloc((size_t)1536 * 512 * 2);
  bf16* rk1t  = (bf16*)alloc((size_t)1536 * 512 * 2);
  bf16* k2t   = (bf16*)alloc((size_t)1536 * 512 * 2);
  bf16* rk2t  = (bf16*)alloc((size_t)1536 * 512 * 2);
  // per-step activations
  bf16* phi1  = (bf16*)alloc((size_t)64 * 256 * 2);
  bf16* phi2  = (bf16*)alloc((size_t)64 * 256 * 2);
  bf16* zp1   = (bf16*)alloc((size_t)64 * 256 * 2);
  bf16* zbuf  = (bf16*)alloc((size_t)64 * 64  * 2);
  bf16* h1b   = (bf16*)alloc((size_t)64 * 512 * 2);
  bf16* h4b   = (bf16*)alloc((size_t)64 * 512 * 2);
  float* pzb  = (float*)alloc((size_t)64 * 128 * 4);
  float* bpz  = (float*)alloc((size_t)128 * 4);
  float* mx1  = (float*)alloc((size_t)64 * 1536 * 4);
  float* mh1  = (float*)alloc((size_t)64 * 1536 * 4);
  float* mx4  = (float*)alloc((size_t)64 * 1536 * 4);
  float* mh4  = (float*)alloc((size_t)64 * 1536 * 4);
  float* h1f  = (float*)alloc((size_t)64 * 512 * 4);
  float* h4f  = (float*)alloc((size_t)64 * 512 * 4);

  float* out    = (float*)d_out;
  float* out_mu = out;
  float* out_sg = out + (size_t)BT * 128;
  float* out_z  = out + (size_t)2 * BT * 128;

  auto cdiv = [](int a, int b) { return (a + b - 1) / b; };

  // ---- weight conversion / transposition (bf16) ----
  conv_bf16<<<cdiv(BT * 128, 256), 256, 0, stream>>>(inX, Xin, BT * 128);
  wconv_t<<<cdiv(128 * 256, 256), 256, 0, stream>>>(Wx1, Wx1t, 128, 256);
  wconv_t<<<cdiv(256 * 256, 256), 256, 0, stream>>>(Wx2, Wx2t, 256, 256);
  wconv_t<<<cdiv(64 * 256, 256),  256, 0, stream>>>(Wz1, Wz1t, 64, 256);
  wconv_t<<<cdiv(256 * 256, 256), 256, 0, stream>>>(Wz2, Wz2t, 256, 256);
  wconv_t<<<cdiv(768 * 256, 256), 256, 0, stream>>>(Wp1, Wp1t, 768, 256);
  wconv_t<<<cdiv(256 * 256, 256), 256, 0, stream>>>(Wp2, Wp2t, 256, 256);
  wconv_t<<<cdiv(256 * 64, 256),  256, 0, stream>>>(Wpmu,  Wpzt,            256, 64);
  wconv_t<<<cdiv(256 * 64, 256),  256, 0, stream>>>(Wpsig, Wpzt + 64 * 256, 256, 64);
  wconv_t<<<cdiv(768 * 256, 256), 256, 0, stream>>>(Wt1, Wt1t, 768, 256);
  wconv_t<<<cdiv(256 * 256, 256), 256, 0, stream>>>(Wt2, Wt2t, 256, 256);
  wconv_t<<<cdiv(256 * 128, 256), 256, 0, stream>>>(Wtmu,  Wtmut,  256, 128);
  wconv_t<<<cdiv(256 * 128, 256), 256, 0, stream>>>(Wtsig, Wtsigt, 256, 128);
  wconv_t<<<cdiv(512 * 1536, 256), 256, 0, stream>>>(k1,  k1t,  512, 1536);
  wconv_t<<<cdiv(512 * 1536, 256), 256, 0, stream>>>(rk1, rk1t, 512, 1536);
  wconv_t<<<cdiv(512 * 1536, 256), 256, 0, stream>>>(k2,  k2t,  512, 1536);
  wconv_t<<<cdiv(512 * 1536, 256), 256, 0, stream>>>(rk2, rk2t, 512, 1536);
  copyf<<<1, 64, 0, stream>>>(bpmu,  bpz,      64);
  copyf<<<1, 64, 0, stream>>>(bpsig, bpz + 64, 64);

  // h0 = 0 (deterministic per launch; graph-capture-safe async memsets)
  hipMemsetAsync(h1f, 0, (size_t)64 * 512 * 4, stream);
  hipMemsetAsync(h4f, 0, (size_t)64 * 512 * 4, stream);
  hipMemsetAsync(h1b, 0, (size_t)64 * 512 * 2, stream);
  hipMemsetAsync(h4b, 0, (size_t)64 * 512 * 2, stream);

  // gemm launcher: D = act(A @ W^T + bias); grid = (M/64)*(N/64) blocks
  auto gemm = [&](const bf16* A0, int lda0, int kc0, const bf16* A1, int lda1,
                  const bf16* Wt, const float* bias,
                  float* Df, int ldDf, bf16* Db, int ldDb,
                  int M, int N, int K, int act, int d0, int d1) {
    int blocks = (M >> 6) * (N >> 6);
    gemm_bf16_wmma<<<blocks, 256, 0, stream>>>(
        A0, lda0, kc0, A1 ? A1 : A0, A1 ? lda1 : lda0, Wt, bias,
        Df, ldDf, Db, ldDb, M, N, K, act, d0, d1);
  };

  // ---- batch-parallel: X = relu(relu(in @ Wx1 + bx1) @ Wx2 + bx2) ----
  gemm(Xin, 128, 128, nullptr, 0, Wx1t, bx1, nullptr, 0, x1b, 256,
       BT, 256, 128, /*relu*/1, 0, 0);
  // remap [B,T] -> [T,B] so per-step slices are contiguous
  gemm(x1b, 256, 256, nullptr, 0, Wx2t, bx2, nullptr, 0, Xbf, 256,
       BT, 256, 256, 1, /*d0=B*/64, /*d1=T*/256);

  // ---- sequential scan over T ----
  for (int t = 0; t < T; ++t) {
    const bf16* Xt  = Xbf + (size_t)t * B * 256;
    bf16*       zpt = zpA + (size_t)t * B * 256;
    bf16*       St  = SA  + (size_t)t * B * 512;

    // phi = relu(relu([x_t | h4] @ Wp1 + bp1) @ Wp2 + bp2)
    gemm(Xt, 256, 256, h4b, 512, Wp1t, bp1, nullptr, 0, phi1, 256,
         64, 256, 768, 1, 0, 0);
    gemm(phi1, 256, 256, nullptr, 0, Wp2t, bp2, nullptr, 0, phi2, 256,
         64, 256, 256, 1, 0, 0);
    // [pmu | psig_pre] = phi @ [Wpmu;Wpsig] + [bpmu;bpsig]
    gemm(phi2, 256, 256, nullptr, 0, Wpzt, bpz, pzb, 128, nullptr, 0,
         64, 128, 256, 0, 0, 0);
    // z_t = pmu + softplus(psig_pre) * eps_t
    z_elem<<<16, 256, 0, stream>>>(pzb, eps, out_z, zbuf, t);
    // zp = relu(relu(z @ Wz1 + bz1) @ Wz2 + bz2)
    gemm(zbuf, 64, 64, nullptr, 0, Wz1t, bz1, nullptr, 0, zp1, 256,
         64, 256, 64, 1, 0, 0);
    gemm(zp1, 256, 256, nullptr, 0, Wz2t, bz2, nullptr, 0, zpt, 256,
         64, 256, 256, 1, 0, 0);
    // GRU1: mx = [x_t | zp] @ k1 + b1[0]; mh = h1 @ rk1 + b1[1]
    gemm(Xt, 256, 256, zpt, 256, k1t, b1, mx1, 1536, nullptr, 0,
         64, 1536, 512, 0, 0, 0);
    gemm(h1b, 512, 512, nullptr, 0, rk1t, b1 + 1536, mh1, 1536, nullptr, 0,
         64, 1536, 512, 0, 0, 0);
    gru_combine<<<128, 256, 0, stream>>>(mx1, mh1, h1f, h1b, nullptr);
    // GRU4: mh4 uses old h4; mx4 uses fresh s1 (== new h1)
    gemm(h4b, 512, 512, nullptr, 0, rk2t, b2 + 1536, mh4, 1536, nullptr, 0,
         64, 1536, 512, 0, 0, 0);
    gemm(h1b, 512, 512, nullptr, 0, k2t, b2, mx4, 1536, nullptr, 0,
         64, 1536, 512, 0, 0, 0);
    gru_combine<<<128, 256, 0, stream>>>(mx4, mh4, h4f, h4b, St);
  }

  // ---- batch-parallel decoder: th = relu(relu([zp|S] @ Wt1) @ Wt2) ----
  gemm(zpA, 256, 256, SA, 512, Wt1t, bt1, nullptr, 0, th1, 256,
       BT, 256, 768, 1, 0, 0);
  gemm(th1, 256, 256, nullptr, 0, Wt2t, bt2, nullptr, 0, th2, 256,
       BT, 256, 256, 1, 0, 0);
  // outputs remap [T,B] -> [B,T]
  gemm(th2, 256, 256, nullptr, 0, Wtmut, btmu, out_mu, 128, nullptr, 0,
       BT, 128, 256, 0, /*d0=T*/256, /*d1=B*/64);
  gemm(th2, 256, 256, nullptr, 0, Wtsigt, btsig, out_sg, 128, nullptr, 0,
       BT, 128, 256, /*softplus*/2, 256, 64);
}
